// MemoryKeyVector_74526272520994
// MI455X (gfx1250) — matrix operand order repaired and checked by reading, not compiled, not run on hardware
//
#include <hip/hip_runtime.h>
#include <cstdint>

// ---------------------------------------------------------------------------
// MemoryKeyVector lookup for MI455X (gfx1250, wave32, WMMA).
//   distances = keys[4096,512] @ memory_keys[65536,512]^T
//   idx = argmax_s distances ; out = memory[idx]  (4096 x 1024 f32)
//
// Strategy: fused tiled GEMM + running argmax using bf16x3 split
// (hi*hi + hi*lo + lo*hi) on V_WMMA_F32_16X16X32_BF16 for ~fp32-accurate
// argmax at bf16 matrix-core rates. B (memory_keys) is pre-converted into
// WMMA B-fragment layout in d_ws so fragment loads are two b128 per lane
// and the 128MB hi+lo working set lives in the 192MB L2.
//
// d_ws layout (needs ~130 MiB):
//   [0)            bhi : 16777216 u32  (64 MiB)  B hi fragments
//   [64MiB)        blo : 16777216 u32  (64 MiB)  B lo fragments
//   [128MiB)       pmax: 64*4096 f32   ( 1 MiB)  per-(chunk,row) partial max
//   [129MiB)       pidx: 64*4096 i32   ( 1 MiB)  per-(chunk,row) partial idx
// ---------------------------------------------------------------------------

typedef __attribute__((ext_vector_type(16))) __bf16 v16bf;
typedef __attribute__((ext_vector_type(8)))  float  v8f;

#define KDIM   512
#define NSLOTS 65536
#define BROWS  4096
#define MSIZE  1024

__device__ __forceinline__ unsigned short f2bf_rne(float x) {
    unsigned u = __float_as_uint(x);
    unsigned r = 0x7FFFu + ((u >> 16) & 1u);   // round to nearest even
    return (unsigned short)((u + r) >> 16);
}
__device__ __forceinline__ float bf2f(unsigned short h) {
    return __uint_as_float(((unsigned)h) << 16);
}

union Frag { uint4 q[2]; v16bf v; };

// ---------------------------------------------------------------------------
// Kernel 1: convert memory_keys (f32, row-major [s][k]) into B-fragment-layout
// hi/lo bf16 arrays. Fragment layout for 16-bit B 32x16 (contiguous K order, same as A):
//   lane l: N = l%16 (slot within 16-tile), half = l/16 selects K in
//   [16*half, 16*half+16); dword v in 0..7 holds K = 16*half + 2v, +1.
// Linear dword index i = ((gt*16 + kslice)*32 + lane)*8 + v, gt = slot/16.
// ---------------------------------------------------------------------------
__global__ void mkv_convert_b_kernel(const float* __restrict__ mkeys,
                                     unsigned* __restrict__ bhi,
                                     unsigned* __restrict__ blo) {
    unsigned i    = blockIdx.x * 256u + threadIdx.x;   // 0 .. 16777215
    unsigned v    = i & 7u;
    unsigned lane = (i >> 3) & 31u;
    unsigned ks   = (i >> 8) & 15u;
    unsigned gt   = i >> 12;
    unsigned n    = lane & 15u;
    unsigned half = lane >> 4;
    unsigned k    = ks * 32u + half * 16u + v * 2u;
    unsigned s    = gt * 16u + n;

    float a0 = mkeys[s * KDIM + k];
    float a1 = mkeys[s * KDIM + k + 1];
    unsigned short h0 = f2bf_rne(a0), h1 = f2bf_rne(a1);
    unsigned short l0 = f2bf_rne(a0 - bf2f(h0));
    unsigned short l1 = f2bf_rne(a1 - bf2f(h1));
    bhi[i] = (unsigned)h0 | ((unsigned)h1 << 16);
    blo[i] = (unsigned)l0 | ((unsigned)l1 << 16);
}

// ---------------------------------------------------------------------------
// Kernel 2: fused GEMM + argmax partials.
// grid = (128 m-blocks of 32 rows, 64 s-chunks of 1024 slots), block = 256.
// Stage A (32x512) as hi/lo bf16 in A-fragment layout in LDS, then 8 waves
// (2 m-tiles x 4 n-subchunks of 256 slots) run 16x16 WMMA tiles over K=512,
// keeping a per-lane running (max, slot) using the C/D layout (lane = col).
// ---------------------------------------------------------------------------
__global__ __launch_bounds__(256) void mkv_argmax_partial_kernel(
    const float* __restrict__ keys,
    const unsigned* __restrict__ bhi,
    const unsigned* __restrict__ blo,
    float* __restrict__ pmax,
    int*   __restrict__ pidx) {

    __shared__ __align__(16) unsigned ldsAhi[8192];  // 2 mt * 16 ks * 32 lane * 8 dw
    __shared__ __align__(16) unsigned ldsAlo[8192];
    __shared__ float sMax[8][16];
    __shared__ int   sIdx[8][16];

    const unsigned tid    = threadIdx.x;
    const unsigned mbase  = blockIdx.x * 32u;
    const unsigned schunk = blockIdx.y;

    // --- Stage A into LDS in 16-bit A 16x32 fragment layout (contiguous K order):
    // lane l: M = l%16, half = l/16; dword v -> K = 16*half + 2v, +1
    // (same K order as the B fragments).
    for (unsigned j = tid; j < 8192u; j += 256u) {
        unsigned v    = j & 7u;
        unsigned lane = (j >> 3) & 31u;
        unsigned ks   = (j >> 8) & 15u;
        unsigned mt   = j >> 12;
        unsigned m    = lane & 15u;
        unsigned h    = lane >> 4;
        unsigned kb   = 16u * h + 2u * v;
        unsigned k    = ks * 32u + kb;
        unsigned row  = mbase + mt * 16u + m;
        float a0 = keys[row * KDIM + k];
        float a1 = keys[row * KDIM + k + 1];
        unsigned short h0 = f2bf_rne(a0), h1 = f2bf_rne(a1);
        unsigned short l0 = f2bf_rne(a0 - bf2f(h0));
        unsigned short l1 = f2bf_rne(a1 - bf2f(h1));
        ldsAhi[j] = (unsigned)h0 | ((unsigned)h1 << 16);
        ldsAlo[j] = (unsigned)l0 | ((unsigned)l1 << 16);
    }
    __syncthreads();

    const unsigned wv   = tid >> 5;   // wave 0..7
    const unsigned lane = tid & 31u;
    const unsigned mt   = wv & 1u;    // m-tile 0..1
    const unsigned nw   = wv >> 1;    // n-subchunk 0..3 (256 slots each)

    float best[8];
    int   bcol[8];
#pragma unroll
    for (int v = 0; v < 8; ++v) { best[v] = -3.4e38f; bcol[v] = 0; }

    for (unsigned ntile = 0; ntile < 16u; ++ntile) {
        const unsigned stile = schunk * 1024u + nw * 256u + ntile * 16u;
        const unsigned gt    = stile >> 4;
        v8f acc = {};
#pragma unroll 4
        for (unsigned ks = 0; ks < 16u; ++ks) {
            const unsigned abase = ((mt * 16u + ks) * 32u + lane) * 8u;
            const unsigned bbase = ((gt * 16u + ks) * 32u + lane) * 8u;
            Frag ah, al, bh, bl;
            ah.q[0] = *(const uint4*)&ldsAhi[abase];
            ah.q[1] = *(const uint4*)&ldsAhi[abase + 4u];
            al.q[0] = *(const uint4*)&ldsAlo[abase];
            al.q[1] = *(const uint4*)&ldsAlo[abase + 4u];
            bh.q[0] = *(const uint4*)&bhi[bbase];
            bh.q[1] = *(const uint4*)&bhi[bbase + 4u];
            bl.q[0] = *(const uint4*)&blo[bbase];
            bl.q[1] = *(const uint4*)&blo[bbase + 4u];
            // bf16x3 split: hi*hi + hi*lo + lo*hi, f32 accumulate
            acc = __builtin_amdgcn_wmma_f32_16x16x32_bf16(false, ah.v, false, bh.v,
                                                          (short)0, acc, false, false);
            acc = __builtin_amdgcn_wmma_f32_16x16x32_bf16(false, ah.v, false, bl.v,
                                                          (short)0, acc, false, false);
            acc = __builtin_amdgcn_wmma_f32_16x16x32_bf16(false, al.v, false, bh.v,
                                                          (short)0, acc, false, false);
        }
        // C/D layout: lane = column N (slot stile + lane%16), VGPR v = row
        const int col = (int)(stile + (lane & 15u));
#pragma unroll
        for (int v = 0; v < 8; ++v) {
            float a = acc[v];
            if (a > best[v]) { best[v] = a; bcol[v] = col; }
        }
    }

    // Cross-lane argmax within each 16-lane half (rows v and v+8).
#pragma unroll
    for (int v = 0; v < 8; ++v) {
#pragma unroll
        for (int off = 8; off >= 1; off >>= 1) {
            float ov = __shfl_xor(best[v], off, 32);
            int   oc = __shfl_xor(bcol[v], off, 32);
            if (ov > best[v] || (ov == best[v] && oc < bcol[v])) {
                best[v] = ov; bcol[v] = oc;
            }
        }
        if ((lane & 15u) == 0u) {
            unsigned r = (unsigned)v + 8u * (lane >> 4);
            sMax[wv][r] = best[v];
            sIdx[wv][r] = bcol[v];
        }
    }
    __syncthreads();

    // Cross-wave argmax: 4 n-waves share each m-tile.
    if (tid < 32u) {
        unsigned rloc = tid & 15u;
        unsigned mtl  = tid >> 4;
        float bv = -3.4e38f;
        int   bi = 0;
        for (unsigned q = 0; q < 4u; ++q) {
            unsigned w = q * 2u + mtl;
            float vq = sMax[w][rloc];
            int   iq = sIdx[w][rloc];
            if (vq > bv || (vq == bv && iq < bi)) { bv = vq; bi = iq; }
        }
        unsigned grow = mbase + mtl * 16u + rloc;
        pmax[schunk * 4096u + grow] = bv;
        pidx[schunk * 4096u + grow] = bi;
    }
}

// ---------------------------------------------------------------------------
// Kernel 3: per-row reduce of 64 chunk partials + gather memory[idx] row.
// grid = 4096 blocks x 256 threads; each thread copies one float4.
// ---------------------------------------------------------------------------
__global__ void mkv_reduce_gather_kernel(const float* __restrict__ pmax,
                                         const int*   __restrict__ pidx,
                                         const float* __restrict__ memory,
                                         float*       __restrict__ out) {
    __shared__ float sv[64];
    __shared__ int   si[64];
    __shared__ int   bestIdx;
    const unsigned b   = blockIdx.x;
    const unsigned tid = threadIdx.x;
    if (tid < 64u) {
        sv[tid] = pmax[tid * 4096u + b];
        si[tid] = pidx[tid * 4096u + b];
    }
    __syncthreads();
    if (tid == 0u) {
        float bv = sv[0]; int bi = si[0];
        for (int t = 1; t < 64; ++t) {
            if (sv[t] > bv || (sv[t] == bv && si[t] < bi)) { bv = sv[t]; bi = si[t]; }
        }
        bestIdx = bi;
    }
    __syncthreads();
    const float4* src = (const float4*)(memory + (size_t)bestIdx * MSIZE);
    float4*       dst = (float4*)(out + (size_t)b * MSIZE);
    dst[tid] = src[tid];
}

// ---------------------------------------------------------------------------
extern "C" void kernel_launch(void* const* d_in, const int* in_sizes, int n_in,
                              void* d_out, int out_size, void* d_ws, size_t ws_size,
                              hipStream_t stream) {
    (void)in_sizes; (void)n_in; (void)out_size; (void)ws_size;
    const float* keys   = (const float*)d_in[0];   // [4096, 512]
    const float* memory = (const float*)d_in[1];   // [65536, 1024]
    const float* mkeys  = (const float*)d_in[2];   // [65536, 512]
    float* out = (float*)d_out;                    // [4096, 1024]

    const size_t B_DWORDS = (size_t)NSLOTS * KDIM / 2;   // 16777216
    unsigned* bhi  = (unsigned*)d_ws;
    unsigned* blo  = bhi + B_DWORDS;
    float*    pmax = (float*)(blo + B_DWORDS);
    int*      pidx = (int*)(pmax + 64 * BROWS);

    // 1) memory_keys -> hi/lo bf16 B-fragments (one dword per thread)
    mkv_convert_b_kernel<<<65536, 256, 0, stream>>>(mkeys, bhi, blo);
    // 2) fused WMMA GEMM + argmax partials
    mkv_argmax_partial_kernel<<<dim3(128, 64), 256, 0, stream>>>(keys, bhi, blo,
                                                                 pmax, pidx);
    // 3) reduce partials per row + gather memory rows
    mkv_reduce_gather_kernel<<<4096, 256, 0, stream>>>(pmax, pidx, memory, out);
}
